// PPGtoECG_13683765805606
// MI455X (gfx1250) — matrix operand order repaired and checked by reading, not compiled, not run on hardware
//
#include <hip/hip_runtime.h>
#include <hip/hip_bf16.h>
#include <math.h>

typedef __attribute__((ext_vector_type(16))) _Float16 v16h;
typedef __attribute__((ext_vector_type(8)))  _Float16 v8h;
typedef __attribute__((ext_vector_type(8)))  float    v8f;
typedef __attribute__((ext_vector_type(4)))  int      v4i;

#define WMMA_F16(a,b,c) __builtin_amdgcn_wmma_f32_16x16x32_f16(false,(a),false,(b),(short)0,(c),false,false)

static __device__ __forceinline__ v16h cat8(v8h lo, v8h hi){
  return __builtin_shufflevector(lo,hi,0,1,2,3,4,5,6,7,8,9,10,11,12,13,14,15);
}
static __device__ __forceinline__ float sigmf(float x){ return 1.0f/(1.0f+__expf(-x)); }

// ---------------- gfx1250 async global->LDS copy (ASYNCcnt-tracked DMA) ----
#if __has_builtin(__builtin_amdgcn_global_load_async_to_lds_b128)
static __device__ __forceinline__ void async_g2l_b128(const void* g, void* l){
  __builtin_amdgcn_global_load_async_to_lds_b128(
      (__attribute__((address_space(1))) v4i*)g,
      (__attribute__((address_space(3))) v4i*)l, 0, 0);
}
#else
static __device__ __forceinline__ void async_g2l_b128(const void* g, void* l){
  unsigned loff = (unsigned)(size_t)(__attribute__((address_space(3))) char*)l;
  asm volatile("global_load_async_to_lds_b128 %0, %1, off"
               :: "v"(loff), "v"(g) : "memory");
}
#endif

#if __has_builtin(__builtin_amdgcn_s_wait_asynccnt)
static __device__ __forceinline__ void wait_async3(){ __builtin_amdgcn_s_wait_asynccnt(3); }
static __device__ __forceinline__ void wait_async0(){ __builtin_amdgcn_s_wait_asynccnt(0); }
#else
static __device__ __forceinline__ void wait_async3(){ asm volatile("s_wait_asynccnt 3" ::: "memory"); }
static __device__ __forceinline__ void wait_async0(){ asm volatile("s_wait_asynccnt 0" ::: "memory"); }
#endif

constexpr float PIF    = 3.14159265358979323846f;
constexpr float TWO_PI = 6.28318530717958647692f;

__constant__ float PLIMS[23][2] = {
  {-1.7f, 1.7f},
  {-70.0f*PIF/180.0f, -50.0f*PIF/180.0f},
  {0.24f, 0.35f},
  {-6.5f, -1.5f},
  {-25.0f*PIF/180.0f, -5.0f*PIF/180.0f},
  {0.08f, 0.12f},
  {15.0f, 55.0f},
  {-20.0f*PIF/180.0f, 20.0f*PIF/180.0f},
  {0.09f, 0.11f},
  {-0.8f, -0.3f},
  {5.0f*PIF/180.0f, 25.0f*PIF/180.0f},
  {1e-16f, 0.1f},
  {0.5f, 0.9f},
  {80.0f*PIF/180.0f, 120.0f*PIF/180.0f},
  {0.3f, 0.5f},
  {0.2f, 0.9f},
  {130.0f*PIF/180.0f, 150.0f*PIF/180.0f},
  {0.15f, 0.25f},
  {0.08f, 0.12f},
  {0.008f, 0.012f},
  {0.23f, 0.27f},
  {0.008f, 0.012f},
  {0.4f, 0.6f}
};
__constant__ float SLIMS[2][2] = { {-PIF, PIF}, {-4e-6f, 4e-6f} };

// ---------------------------------------------------------------- f32 -> f16
__global__ void k_cvt(const float* __restrict__ s, _Float16* __restrict__ d, long n){
  long i = (long)blockIdx.x*blockDim.x + threadIdx.x;
  long st = (long)gridDim.x*blockDim.x;
  for (; i < n; i += st) d[i] = (_Float16)s[i];
}

// ------------------------------------------------- big parallel WMMA GEMM
// C[M][N] (f16) = A[M][K](f16) @ W[N][K]^T (f16) + bias[N]
// block = 8 waves, tile 128 x 64, K chunked by 32.
// A/B tiles staged into double-buffered LDS with async global->LDS DMA;
// ASYNCcnt pipelining: while computing chunk kc, chunk kc+1 is in flight.
__global__ void __launch_bounds__(256)
k_gemm_f16(const _Float16* __restrict__ A, const _Float16* __restrict__ W,
           const float* __restrict__ bias, _Float16* __restrict__ C,
           int M, int N, int K){
  __shared__ __align__(16) _Float16 Abuf[2][128*32];   // 8 KB each
  __shared__ __align__(16) _Float16 Bbuf[2][64*32];    // 4 KB each
  const int tid = threadIdx.x;
  const int wv = tid >> 5, lane = tid & 31;
  const int l15 = lane & 15, g2 = lane >> 4;
  const int Mbase = blockIdx.x*128;
  const int Nb = blockIdx.y*64;
  const int nk = K >> 5;

  // stage chunk kc into buffer q: 3 async-to-LDS instructions per thread
  auto issue = [&](int kc, int q){
    #pragma unroll
    for (int i = 0; i < 2; ++i){
      const int idx = tid + i*256;           // 0..511
      const int row = idx >> 2, seg = idx & 3;
      async_g2l_b128(A + (size_t)(Mbase + row)*K + kc*32 + seg*8,
                     &Abuf[q][row*32 + seg*8]);
    }
    {
      const int row = tid >> 2, seg = tid & 3;
      async_g2l_b128(W + (size_t)(Nb + row)*K + kc*32 + seg*8,
                     &Bbuf[q][row*32 + seg*8]);
    }
  };

  issue(0, 0);
  if (nk > 1) issue(1, 1);

  v8f zero = {};
  v8f acc[4] = {zero, zero, zero, zero};
  for (int kc = 0; kc < nk; ++kc){
    const int q = kc & 1;
    if (kc + 1 < nk) wait_async3(); else wait_async0();
    __syncthreads();
    const int ar = (wv*16 + l15)*32;
    v16h a = cat8(*(const v8h*)(Abuf[q] + ar + g2*16),
                  *(const v8h*)(Abuf[q] + ar + g2*16 + 8));
    #pragma unroll
    for (int nt = 0; nt < 4; ++nt){
      v16h b = *(const v16h*)(Bbuf[q] + (nt*16 + l15)*32 + g2*16);
      acc[nt] = WMMA_F16(a, b, acc[nt]);
    }
    __syncthreads();                          // all reads of buf q done
    if (kc + 2 < nk) issue(kc + 2, q);        // reuse buf q
  }
  const int Mt = Mbase + wv*16;
  #pragma unroll
  for (int nt = 0; nt < 4; ++nt){
    const int n = Nb + nt*16 + l15;
    const float bv = bias[n];
    #pragma unroll
    for (int r = 0; r < 8; ++r){
      const int m = Mt + r + g2*8;
      C[(size_t)m*N + n] = (_Float16)(acc[nt][r] + bv);
    }
  }
}

// ------------------------------------------------- persistent LSTM scan
// One block = 16 batch rows (blockIdx.x*16). 16 waves (512 threads):
//   wave w (w<16):  hidden block j in [16w,16w+16), K-half 0..127, owns i/f/g/o
//   no wave w+16: kw = wv>>4 is always 0, so K-half 128..255 is never computed
// Whh (f16 [1024][256]) held register-resident as WMMA B fragments for the
// entire scan: 4 gates x 4 K-chunks x v16h = 128 VGPRs/wave.
__global__ void __launch_bounds__(512)
k_lstm_scan(const float* __restrict__ xsmall, int xK, int xmode,
            const float* __restrict__ Wih, const float* __restrict__ bias,
            const _Float16* __restrict__ gates_in,
            const _Float16* __restrict__ Whh,
            _Float16* __restrict__ h_out, long hout_stride, int hcol_off,
            float* __restrict__ c_out, long cout_stride,
            int T, int Ball, int reverse)
{
  __shared__ __align__(16) _Float16 hsh[16*256];   // current hidden state (f16)
  __shared__ float partial[16*512];                // K-partial exchange (2 gates)
  const int tid = threadIdx.x;
  const int wv = tid >> 5, lane = tid & 31;
  const int kw = wv >> 4;            // which K half
  const int hb = wv & 15;            // hidden block
  const int l15 = lane & 15, g2 = lane >> 4;
  const int bb = blockIdx.x*16;

  // load register-resident Whh B-fragments: B[k][n] = Whh[n*256 + k]
  v16h Bf[4][4];
  #pragma unroll
  for (int g = 0; g < 4; ++g){
    const int n = g*256 + hb*16 + l15;
    #pragma unroll
    for (int kc = 0; kc < 4; ++kc){
      const int k0 = kw*128 + kc*32 + g2*16;
      Bf[g][kc] = *(const v16h*)(Whh + (size_t)n*256 + k0);
    }
  }
  float creg[8];
  #pragma unroll
  for (int r = 0; r < 8; ++r) creg[r] = 0.0f;
  for (int i = tid; i < 16*256; i += 512) hsh[i] = (_Float16)0.0f;
  __syncthreads();

  for (int stp = 0; stp < T; ++stp){
    const int t = reverse ? (T-1-stp) : stp;
    v8f zero = {};
    v8f acc[4] = {zero, zero, zero, zero};
    #pragma unroll
    for (int kc = 0; kc < 4; ++kc){
      const int kb = kw*128 + kc*32;
      v16h a = cat8(*(const v8h*)(hsh + l15*256 + kb + g2*16),
                    *(const v8h*)(hsh + l15*256 + kb + g2*16 + 8));
      #pragma unroll
      for (int g = 0; g < 4; ++g) acc[g] = WMMA_F16(a, Bf[g][kc], acc[g]);
    }
    // phase 0: exchange K-partials for gates i,f
    if (kw){
      #pragma unroll
      for (int g = 0; g < 2; ++g)
        #pragma unroll
        for (int r = 0; r < 8; ++r)
          partial[hb*512 + g*256 + (r + g2*8)*16 + l15] = acc[g][r];
    }
    __syncthreads();
    if (!kw){
      #pragma unroll
      for (int g = 0; g < 2; ++g)
        #pragma unroll
        for (int r = 0; r < 8; ++r)
          acc[g][r] += partial[hb*512 + g*256 + (r + g2*8)*16 + l15];
    }
    __syncthreads();
    // phase 1: gates g,o
    if (kw){
      #pragma unroll
      for (int g = 2; g < 4; ++g)
        #pragma unroll
        for (int r = 0; r < 8; ++r)
          partial[hb*512 + (g-2)*256 + (r + g2*8)*16 + l15] = acc[g][r];
    }
    __syncthreads();
    if (!kw){
      const int j = hb*16 + l15;
      #pragma unroll
      for (int r = 0; r < 8; ++r){
        const int m = r + g2*8;
        const int b = bb + m;
        float pre0 = acc[0][r];
        float pre1 = acc[1][r];
        float pre2 = acc[2][r] + partial[hb*512 + 0*256 + m*16 + l15];
        float pre3 = acc[3][r] + partial[hb*512 + 1*256 + m*16 + l15];
        if (gates_in){
          const _Float16* gp = gates_in + ((size_t)t*Ball + b)*1024 + j;
          pre0 += (float)gp[0];   pre1 += (float)gp[256];
          pre2 += (float)gp[512]; pre3 += (float)gp[768];
          if (stp + 1 < T){
            const int tn = reverse ? (t-1) : (t+1);
            __builtin_prefetch(gates_in + ((size_t)tn*Ball + b)*1024 + j, 0, 3);
          }
        } else {
          float q0 = bias[j], q1 = bias[256+j], q2 = bias[512+j], q3 = bias[768+j];
          for (int k = 0; k < xK; ++k){
            const float xv = xmode ? xsmall[((size_t)t*Ball + b)*xK + k]
                                   : xsmall[((size_t)b*xK + k)*T + t];
            q0 += xv * Wih[(size_t)(      j)*xK + k];
            q1 += xv * Wih[(size_t)(256 + j)*xK + k];
            q2 += xv * Wih[(size_t)(512 + j)*xK + k];
            q3 += xv * Wih[(size_t)(768 + j)*xK + k];
          }
          pre0 += q0; pre1 += q1; pre2 += q2; pre3 += q3;
        }
        const float ig = sigmf(pre0), fg = sigmf(pre1);
        const float gg = tanhf(pre2), og = sigmf(pre3);
        const float cc = fg*creg[r] + ig*gg;
        creg[r] = cc;
        const float hh = og * tanhf(cc);
        hsh[m*256 + j] = (_Float16)hh;
        if (h_out) h_out[((size_t)t*Ball + b)*hout_stride + hcol_off + j] = (_Float16)hh;
      }
    }
    __syncthreads();
  }
  if (!kw && c_out){
    const int j = hb*16 + l15;
    #pragma unroll
    for (int r = 0; r < 8; ++r){
      const int b = bb + r + g2*8;
      c_out[(size_t)b*cout_stride + j] = creg[r];
    }
  }
}

// ------------------------------------------------- projection + decode + jacobian
__global__ void __launch_bounds__(256)
k_proj_decode(const float* __restrict__ cells, const float* __restrict__ projW,
              const float* __restrict__ projb, const float* __restrict__ mixw,
              float* __restrict__ param_out, float* __restrict__ state0,
              float* __restrict__ Tn, float* __restrict__ jac_out)
{
  __shared__ float zsh[64*25];
  __shared__ float psh[64*23];
  __shared__ float stsh[64*2];
  __shared__ float tnsh[64*16];
  __shared__ float red[3][64];
  const int tid = threadIdx.x;
  for (int idx = tid; idx < 64*25; idx += 256){
    const int b = idx/25, c = idx%25;
    const float* cr = cells + b*1024;
    const float* wr = projW + c*1024;
    float s = projb[c];
    for (int k = 0; k < 1024; ++k) s += cr[k]*wr[k];
    zsh[idx] = s;
  }
  __syncthreads();
  float w0 = mixw[0], w1 = mixw[1], w2 = mixw[2];
  const float mx = fmaxf(w0, fmaxf(w1, w2));
  const float e0 = __expf(w0-mx), e1 = __expf(w1-mx), e2 = __expf(w2-mx);
  const float es = e0+e1+e2;
  w0 = e0/es; w1 = e1/es; w2 = e2/es;
  for (int idx = tid; idx < 64*25; idx += 256){
    const int b = idx/25, p = idx%25;
    const float x = zsh[idx];
    const float sp = fmaxf(x, 0.0f) + log1pf(__expf(-fabsf(x)));
    const float f1 = sp/(1.0f + fabsf(x));
    const float f2 = tanhf(x);
    const float f3 = 2.0f*sigmf(x) - 1.0f;
    const float mix = w0*f1 + w1*f2 + w2*f3;   // weights sum to 1
    float lo, hi;
    if (p < 23){ lo = PLIMS[p][0];    hi = PLIMS[p][1]; }
    else       { lo = SLIMS[p-23][0]; hi = SLIMS[p-23][1]; }
    const float v = (hi-lo)*(mix+1.0f)*0.5f + lo;
    if (p < 23){ psh[b*23+p] = v; param_out[b*23+p] = v; }
    else stsh[b*2 + (p-23)] = v;
  }
  __syncthreads();
  for (int b = tid; b < 64; b += 256){
    const float s0 = stsh[b*2], s1 = stsh[b*2+1];
    state0[b*3+0] = cosf(s0); state0[b*3+1] = sinf(s0); state0[b*3+2] = s1;
  }
  // Tn via real part of 16-point inverse DFT
  for (int idx = tid; idx < 64*16; idx += 256){
    const int b = idx/16, n = idx%16;
    const float m1 = psh[b*23+18], sd1 = psh[b*23+19];
    const float m2 = psh[b*23+20], sd2 = psh[b*23+21], a2 = psh[b*23+22];
    float acc = 0.0f;
    for (int k = 0; k < 16; ++k){
      const float fk = 0.5f*(float)k/15.0f;
      const float g1  = (1.0f/(sd1*1.41421356237f))*__expf(-0.5f*(fk-m1)/sd1);
      const float g2v = (1.0f/(sd2*1.41421356237f))*__expf(-0.5f*(fk-m2)/sd2);
      const float S = g1 + g2v*a2;
      const float ph = TWO_PI*(float)k/15.0f;
      acc += sqrtf(S)*cosf(ph + TWO_PI*(float)k*(float)n/16.0f);
    }
    const float v = acc/16.0f;
    tnsh[idx] = v; Tn[idx] = v;
  }
  __syncthreads();
  // analytic jacobian Frobenius norms of _dyn at (state0, param, Tn[:,0])
  for (int b = tid; b < 64; b += 256){
    const float xx = cosf(stsh[b*2]);
    const float yy = sinf(stsh[b*2]);
    const float r = sqrtf(xx*xx + yy*yy);
    const float alpha = 1.0f - r;
    const float theta = atan2f(yy, xx);
    const float Tcol = tnsh[b*16 + 0];
    const float om = TWO_PI / Tcol;
    const float dxdx = alpha - xx*xx/r;
    const float dxdy = -xx*yy/r - om;
    const float dydx = -yy*xx/r + om;
    const float dydy = alpha - yy*yy/r;
    float P = 0.0f, Fp = 0.0f;
    for (int jj = 0; jj < 6; ++jj){
      const float aj  = psh[b*23 + jj*3 + 0];
      const float tj  = psh[b*23 + jj*3 + 1];
      const float bwj = psh[b*23 + jj*3 + 2];
      float dth = theta - tj;
      dth -= rintf(dth/TWO_PI)*TWO_PI;
      const float E = __expf(-dth*dth/(2.0f*bwj*bwj));
      const float dterm = E*(1.0f - dth*dth/(bwj*bwj));
      P += aj*dterm;
      const float dda = -dth*E;
      const float ddt = aj*dterm;
      const float ddb = -aj*dth*dth*dth*E/(bwj*bwj*bwj);
      Fp += dda*dda + ddt*ddt + ddb*ddb;
    }
    const float dzdth = -P;
    const float dzdx = dzdth * (-yy/(r*r));
    const float dzdy = dzdth * ( xx/(r*r));
    const float Fs = dxdx*dxdx + dxdy*dxdy + dydx*dydx + dydy*dydy
                   + dzdx*dzdx + dzdy*dzdy + 1.0f;
    const float c2 = TWO_PI/(Tcol*Tcol);
    const float Ft = c2*c2*(xx*xx + yy*yy);
    red[0][b] = Fs; red[1][b] = Fp; red[2][b] = Ft;
  }
  __syncthreads();
  if (tid == 0){
    float sa = 0.0f, sb = 0.0f, sc = 0.0f;
    for (int b = 0; b < 64; ++b){ sa += red[0][b]; sb += red[1][b]; sc += red[2][b]; }
    jac_out[0] = sqrtf(sa) + sqrtf(sb) + sqrtf(sc);
  }
}

// ------------------------------------------------- state-space simulation
__global__ void k_simulate(const float* __restrict__ state0, const float* __restrict__ param,
                           const float* __restrict__ Tn, float* __restrict__ states, int T)
{
  const int b = threadIdx.x;
  if (b >= 64) return;
  float a[6], th[6], bw[6];
  #pragma unroll
  for (int j = 0; j < 6; ++j){
    a[j]  = param[b*23 + j*3 + 0];
    th[j] = param[b*23 + j*3 + 1];
    bw[j] = param[b*23 + j*3 + 2];
  }
  float x = state0[b*3+0], y = state0[b*3+1], z = state0[b*3+2];
  states[(size_t)b*3+0] = x; states[(size_t)b*3+1] = y; states[(size_t)b*3+2] = z;
  int is = 1;
  const float G = __expf(0.1f);
  for (int t = 1; t < T; ++t){
    const int ic = is < 0 ? 0 : (is > 15 ? 15 : is);
    const float Tcol = Tn[b*16 + ic];
    const float r = sqrtf(x*x + y*y);
    const float alpha = 1.0f - r;
    const float theta = atan2f(y, x);
    const float om = TWO_PI/Tcol;
    float dx = alpha*x - om*y;
    float dy = alpha*y + om*x;
    float zs = 0.0f;
    #pragma unroll
    for (int j = 0; j < 6; ++j){
      float dth = theta - th[j];
      dth -= rintf(dth/TWO_PI)*TWO_PI;
      zs += a[j]*dth*__expf(-dth*dth/(2.0f*bw[j]*bw[j]));
    }
    float dz = -zs - z;
    const int close = (fabsf(theta) < 0.0005f) ? 1 : 0;
    const float nn = sqrtf(dx*dx + dy*dy + dz*dz);
    if (nn > 1.0e6f){ const float sc = 1.0e6f/nn; dx *= sc; dy *= sc; dz *= sc; }
    x += dx*G; y += dy*G; z += dz*G;
    states[((size_t)t*64 + b)*3 + 0] = x;
    states[((size_t)t*64 + b)*3 + 1] = y;
    states[((size_t)t*64 + b)*3 + 2] = z;
    is += close;
  }
}

// ------------------------------------------------- output projection
__global__ void k_outproj(const _Float16* __restrict__ d1, const float* __restrict__ W,
                          const float* __restrict__ bb, float* __restrict__ ecg)
{
  const long idx = (long)blockIdx.x*blockDim.x + threadIdx.x;
  if (idx >= 1024L*64*12) return;
  const int j = (int)(idx % 12);
  const long tb = idx / 12;
  const int b = (int)(tb % 64);
  const long t = tb / 64;
  const _Float16* row = d1 + tb*256;
  float s = bb[j];
  for (int k = 0; k < 256; ++k) s += (float)row[k]*W[j*256 + k];
  ecg[((long)b*1024 + t)*12 + j] = s;
}

// ================================================================ host
extern "C" void kernel_launch(void* const* d_in, const int* in_sizes, int n_in,
                              void* d_out, int out_size, void* d_ws, size_t ws_size,
                              hipStream_t stream)
{
  (void)in_sizes; (void)n_in; (void)out_size; (void)ws_size;
  const float* x     = (const float*)d_in[0];
  const float* eWih0 = (const float*)d_in[1];
  const float* eWhh0 = (const float*)d_in[2];
  const float* eb0   = (const float*)d_in[3];
  const float* eWih1 = (const float*)d_in[4];
  const float* eWhh1 = (const float*)d_in[5];
  const float* eb1   = (const float*)d_in[6];
  const float* projW = (const float*)d_in[7];
  const float* projb = (const float*)d_in[8];
  const float* dWih0 = (const float*)d_in[9];
  const float* dWhh0 = (const float*)d_in[10];
  const float* db0   = (const float*)d_in[11];
  const float* dWih1 = (const float*)d_in[12];
  const float* dWhh1 = (const float*)d_in[13];
  const float* db1   = (const float*)d_in[14];
  const float* outW  = (const float*)d_in[15];
  const float* outb  = (const float*)d_in[16];
  const float* mixw  = (const float*)d_in[17];

  const int Tenc = 1000, Tdec = 1024;

  float* ecg    = (float*)d_out;                       // (64,1024,12)
  float* states = ecg + (size_t)64*1024*12;            // (1024,64,3)
  float* param  = states + (size_t)1024*64*3;          // (64,23)
  float* jac    = param + 64*23;                       // scalar

  _Float16* p = (_Float16*)d_ws;
  auto take = [&](size_t n){ _Float16* q = p; p += n; return q; };
  _Float16* wEWhh0 = take((size_t)2*1024*256);
  _Float16* wEWih1 = take((size_t)2*1024*512);
  _Float16* wEWhh1 = take((size_t)2*1024*256);
  _Float16* wDWhh0 = take((size_t)1024*256);
  _Float16* wDWih1 = take((size_t)1024*256);
  _Float16* wDWhh1 = take((size_t)1024*256);
  _Float16* h0     = take((size_t)1000*64*512);        // enc L0 bidir hidden
  _Float16* gatesA = take((size_t)1024*64*1024);       // shared ih-gates buffer
  _Float16* d0b    = take((size_t)1024*64*256);
  _Float16* d1b    = take((size_t)1024*64*256);
  float* fp    = (float*)p;
  float* cells = fp;  fp += 64*1024;
  float* st0   = fp;  fp += 64*3;
  float* TnBuf = fp;  fp += 64*16;

  // weight conversions f32 -> f16
  k_cvt<<<256,256,0,stream>>>(eWhh0, wEWhh0, 2L*1024*256);
  k_cvt<<<256,256,0,stream>>>(eWih1, wEWih1, 2L*1024*512);
  k_cvt<<<256,256,0,stream>>>(eWhh1, wEWhh1, 2L*1024*256);
  k_cvt<<<256,256,0,stream>>>(dWhh0, wDWhh0, 1024L*256);
  k_cvt<<<256,256,0,stream>>>(dWih1, wDWih1, 1024L*256);
  k_cvt<<<256,256,0,stream>>>(dWhh1, wDWhh1, 1024L*256);

  // encoder layer 0 (inline K=12 ih), fwd then bwd
  k_lstm_scan<<<4,512,0,stream>>>(x, 12, 0, eWih0, eb0, nullptr, wEWhh0,
                                  h0, 512, 0, cells + 0, 1024, Tenc, 64, 0);
  k_lstm_scan<<<4,512,0,stream>>>(x, 12, 0, eWih0 + (size_t)4096*12, eb0 + 4096,
                                  nullptr, wEWhh0 + (size_t)1024*256,
                                  h0, 512, 256, cells + 256, 1024, Tenc, 64, 1);
  // encoder layer 1: giant parallel ih GEMM + scan (fwd, then bwd reusing buffer)
  k_gemm_f16<<<dim3(500,16),256,0,stream>>>(h0, wEWih1, eb1, gatesA, 64000, 1024, 512);
  k_lstm_scan<<<4,512,0,stream>>>(nullptr, 0, 0, nullptr, nullptr, gatesA, wEWhh1,
                                  nullptr, 0, 0, cells + 512, 1024, Tenc, 64, 0);
  k_gemm_f16<<<dim3(500,16),256,0,stream>>>(h0, wEWih1 + (size_t)1024*512, eb1 + 4096,
                                            gatesA, 64000, 1024, 512);
  k_lstm_scan<<<4,512,0,stream>>>(nullptr, 0, 0, nullptr, nullptr, gatesA,
                                  wEWhh1 + (size_t)1024*256,
                                  nullptr, 0, 0, cells + 768, 1024, Tenc, 64, 1);
  // projection, bounded decode, Tn, analytic jacobian
  k_proj_decode<<<1,256,0,stream>>>(cells, projW, projb, mixw, param, st0, TnBuf, jac);
  // state simulation (writes states output region)
  k_simulate<<<1,64,0,stream>>>(st0, param, TnBuf, states, Tdec);
  // decoder layer 0 (inline K=3 ih on simulated states)
  k_lstm_scan<<<4,512,0,stream>>>(states, 3, 1, dWih0, db0, nullptr, wDWhh0,
                                  d0b, 256, 0, nullptr, 0, Tdec, 64, 0);
  // decoder layer 1: parallel ih GEMM + scan
  k_gemm_f16<<<dim3(512,16),256,0,stream>>>(d0b, wDWih1, db1, gatesA, 65536, 1024, 256);
  k_lstm_scan<<<4,512,0,stream>>>(nullptr, 0, 0, nullptr, nullptr, gatesA, wDWhh1,
                                  d1b, 256, 0, nullptr, 0, Tdec, 64, 0);
  // output projection -> ecg
  k_outproj<<<(1024*64*12)/256,256,0,stream>>>(d1b, outW, outb, ecg);
}
